// snn_1_6828998001149
// MI455X (gfx1250) — matrix operand order repaired and checked by reading, not compiled, not run on hardware
//
#include <hip/hip_runtime.h>

// ---------- types matching CDNA5 WMMA operand layouts ----------
typedef _Float16 v4h  __attribute__((ext_vector_type(4)));
typedef _Float16 v8h  __attribute__((ext_vector_type(8)));
typedef _Float16 v16h __attribute__((ext_vector_type(16)));
typedef float    v4f  __attribute__((ext_vector_type(4)));
typedef float    v8f  __attribute__((ext_vector_type(8)));
typedef unsigned int u32x4 __attribute__((ext_vector_type(4)));
typedef int          i32x4 __attribute__((ext_vector_type(4)));
typedef int          i32x8 __attribute__((ext_vector_type(8)));

union Frag16 { v16h v; v8h h[2]; };

#if defined(__has_builtin)
#  if __has_builtin(__builtin_amdgcn_tensor_load_to_lds)
#    define HAVE_TDM 1
#  else
#    define HAVE_TDM 0
#  endif
#  if __has_builtin(__builtin_amdgcn_s_wait_tensorcnt)
#    define WAIT_TENSOR() __builtin_amdgcn_s_wait_tensorcnt(0)
#  else
#    define WAIT_TENSOR() asm volatile("s_wait_tensorcnt 0x0" ::: "memory")
#  endif
#else
#  define HAVE_TDM 0
#  define WAIT_TENSOR()
#endif

// ---------------------------------------------------------------
// Prep: f32->f16 weights (W_out zero-padded 18->32 rows), padded
// bias3/beta3/thr3, zero all membrane state.
// ---------------------------------------------------------------
__global__ __launch_bounds__(256) void snn_prep_kernel(
    const float* __restrict__ Wi, const float* __restrict__ Wh,
    const float* __restrict__ Wo, const float* __restrict__ b3,
    const float* __restrict__ be3,
    _Float16* __restrict__ Wi16, _Float16* __restrict__ Wh16,
    _Float16* __restrict__ Wo16,
    float* __restrict__ b3p, float* __restrict__ beta3p,
    float* __restrict__ thr3p,
    float* __restrict__ mem1, float* __restrict__ mem2,
    float* __restrict__ mem3) {
  int i = blockIdx.x * blockDim.x + threadIdx.x;
  if (i < 1024 * 1024) Wi16[i] = (_Float16)Wi[i];
  if (i < 512 * 1024) { Wh16[i] = (_Float16)Wh[i]; mem1[i] = 0.f; }
  if (i < 512 * 512)  mem2[i] = 0.f;
  if (i < 512 * 32)   mem3[i] = 0.f;
  if (i < 32 * 512)   Wo16[i] = (i < 18 * 512) ? (_Float16)Wo[i] : (_Float16)0.f;
  if (i < 32) {
    b3p[i]    = (i < 18) ? b3[i]  : 0.f;
    beta3p[i] = (i < 18) ? be3[i] : 0.f;
    thr3p[i]  = 1.0f;
  }
}

// ---------------------------------------------------------------
// Fused GEMM (f16 WMMA, f32 accum) + LIF update.
// SHAREA: the 4 waves of a block share one 16-row A strip. The
//   Tensor Data Mover stages it into LDS once (D#: 2-D tile 16 x KC,
//   row stride lda). If the source is f32 (CVTA), a one-time
//   cooperative pass converts the raw strip to packed f16 in LDS,
//   so the K-loop always consumes f16 via ds_load_b128.
// B (weights) streams from L2 through a 4-buffer, 2-chunk-deep
//   register pipeline; A fragments are depth-1 from LDS.
// WMMA operand layouts per CDNA5 ISA 7.12.2 (wave32).
// ---------------------------------------------------------------
template <int NT, bool F32OUT, bool CVTA, bool SHAREA, int KC>
__global__ __launch_bounds__(128) void snn_gemm_lif_kernel(
    const _Float16* __restrict__ A,   // [M x KC] f16 (CVTA=false)
    const float* __restrict__ Af,     // [M x lda] f32 (CVTA=true)
    const _Float16* __restrict__ Wt,  // [N x KC] row-major f16
    const float* __restrict__ bias, const float* __restrict__ beta,
    const float* __restrict__ thr,
    float* __restrict__ mem,          // [M x N] persistent state
    _Float16* __restrict__ spkH,      // [M x N] f16 spikes
    float* __restrict__ spkF,         // [M x outN] f32 spikes (final)
    int M, int N, int lda, int outN) {
  // f16 strip region [0, 16*KC*2) ; raw f32 staging region after it
  constexpr int HBYTES = 16 * KC * 2;
  constexpr int ABYTES = SHAREA ? (CVTA ? HBYTES * 3 : HBYTES) : 16;
  __shared__ __align__(16) char ldsA[ABYTES];

  const int lane = threadIdx.x & 31;
  const int wid  = blockIdx.x * (blockDim.x >> 5) + (threadIdx.x >> 5);
  const int nGroups = N / (16 * NT);
  const int mTile = wid / nGroups;
  const int n0    = (wid % nGroups) * 16 * NT;
  const int q     = lane >> 4;
  const int koff  = q * 8;
  const int mRowBase = mTile * 16;

  // ---- stage A strip into LDS (once per block) ----
  if (SHAREA) {
#if HAVE_TDM
    if (threadIdx.x < 32) {  // wave 0 issues the TDM op (EXEC ignored)
      unsigned long long ga = CVTA
          ? (unsigned long long)(uintptr_t)(Af + (size_t)mRowBase * lda)
          : (unsigned long long)(uintptr_t)(A + (size_t)mRowBase * lda);
      unsigned ldsAddr = (unsigned)(uintptr_t)&ldsA[0] + (CVTA ? HBYTES : 0);
      u32x4 g0;
      g0.x = 1u;                                    // count=1, user mode
      g0.y = ldsAddr;                               // lds_addr
      g0.z = (unsigned)(ga & 0xFFFFFFFFu);          // global_addr[31:0]
      g0.w = (unsigned)((ga >> 32) & 0x1FFFFFFu) | (2u << 30);  // type=2
      const int dsz = CVTA ? 2 : 1;                 // 4B : 2B element
      unsigned long long st0 = (unsigned long long)lda;
      i32x8 g1;
      g1[0] = dsz << 16;                            // data_size
      g1[1] = (int)(((unsigned)KC & 0xFFFFu) << 16);        // dim0 lo
      g1[2] = (int)(((unsigned)KC >> 16) | (16u << 16));    // dim0 hi, dim1=16
      g1[3] = (int)(((unsigned)KC & 0xFFFFu) << 16);        // tile_dim0=KC
      g1[4] = 16;                                   // tile_dim1=16, tile_dim2=0
      g1[5] = (int)(st0 & 0xFFFFFFFFu);             // dim0_stride lo
      g1[6] = (int)((st0 >> 32) & 0xFFFFu);         // dim0_stride hi
      g1[7] = 0;
      i32x4 gz = (i32x4)(0);
#if __clang_major__ >= 23
      i32x8 gz8 = (i32x8)(0);
      __builtin_amdgcn_tensor_load_to_lds(g0, g1, gz, gz, gz8, 0);
#else
      __builtin_amdgcn_tensor_load_to_lds(g0, g1, gz, gz, 0);
#endif
      WAIT_TENSOR();
    }
    __syncthreads();
    if (CVTA) {
      // one-time cooperative f32 -> packed f16 conversion in LDS
      const float* srcF = (const float*)(ldsA + HBYTES);
      _Float16*    dstH = (_Float16*)ldsA;
      for (int j = threadIdx.x * 4; j < 16 * KC; j += 128 * 4) {
        v4f f = *(const v4f*)(srcF + j);
        *(v4h*)(dstH + j) = __builtin_convertvector(f, v4h);
      }
      __syncthreads();
    }
#else
    // cooperative fallback copy (convert inline if needed)
    _Float16* dst = (_Float16*)ldsA;
    for (int j = threadIdx.x; j < 16 * KC; j += 128) {
      size_t g = (size_t)(mRowBase + j / KC) * lda + (j % KC);
      dst[j] = CVTA ? (_Float16)Af[g] : A[g];
    }
    __syncthreads();
#endif
  }

  // ---- fragment addressing (A is always f16 when SHAREA) ----
  const int m = lane & 15;
  const _Float16* aRowH = nullptr;
  const float*    aRowF = nullptr;
  if (SHAREA) {
    aRowH = (const _Float16*)ldsA + (size_t)m * KC;
  } else {
    if (CVTA) aRowF = Af + (size_t)(mRowBase + m) * lda;
    else      aRowH = A + (size_t)(mRowBase + m) * lda;
  }
  const _Float16* bRow[NT];
#pragma unroll
  for (int nt = 0; nt < NT; ++nt)
    bRow[nt] = Wt + (size_t)(n0 + nt * 16 + m) * KC + q * 8;

  v8f acc[NT];
#pragma unroll
  for (int nt = 0; nt < NT; ++nt) acc[nt] = (v8f)0.f;

  auto loadA = [&](Frag16& a, int k0) {
    if (!SHAREA && CVTA) {
      v8f f0 = *(const v8f*)(aRowF + k0 + koff);
      v8f f1 = *(const v8f*)(aRowF + k0 + 16 + koff);
      a.h[0] = __builtin_convertvector(f0, v8h);
      a.h[1] = __builtin_convertvector(f1, v8h);
    } else {
      a.h[0] = *(const v8h*)(aRowH + k0 + koff);
      a.h[1] = *(const v8h*)(aRowH + k0 + 16 + koff);
    }
  };
  auto loadB = [&](Frag16* b, int k0) {
#pragma unroll
    for (int nt = 0; nt < NT; ++nt) {
      b[nt].h[0] = *(const v8h*)(bRow[nt] + k0);
      b[nt].h[1] = *(const v8h*)(bRow[nt] + k0 + 16);
    }
  };
  auto mmac = [&](Frag16& a, Frag16* b) {
#pragma unroll
    for (int nt = 0; nt < NT; ++nt)
      acc[nt] = __builtin_amdgcn_wmma_f32_16x16x32_f16(
          false, a.v, false, b[nt].v, (short)0, acc[nt], false, false);
  };

  // -------- main loop: B depth-2 (4 buffers), A depth-1 ----------
  Frag16 a0, a1, b0[NT], b1[NT], b2[NT], b3[NT];
  loadB(b0, 0);
  loadB(b1, 32);
  loadA(a0, 0);
  int k = 0;
#pragma unroll 1
  for (; k + 128 < KC; k += 128) {
    loadB(b2, k + 64);
    loadB(b3, k + 96);
    loadA(a1, k + 32);
    mmac(a0, b0);
    loadA(a0, k + 64);
    mmac(a1, b1);
    loadB(b0, k + 128);
    loadB(b1, k + 160);
    loadA(a1, k + 96);
    mmac(a0, b2);
    loadA(a0, k + 128);
    mmac(a1, b3);
  }
  loadB(b2, k + 64);
  loadB(b3, k + 96);
  loadA(a1, k + 32);
  mmac(a0, b0);
  loadA(a0, k + 64);
  mmac(a1, b1);
  loadA(a1, k + 96);
  mmac(a0, b2);
  mmac(a1, b3);

  // ---- fused LIF epilogue (snntorch Leaky, reset-by-subtraction) ----
  const int rowBase = mRowBase + q * 8;
#pragma unroll
  for (int nt = 0; nt < NT; ++nt) {
    const int col = n0 + nt * 16 + m;
    const float bcol  = bias[col];
    const float betac = beta[col];
    const float thrc  = thr[col];
#pragma unroll
    for (int r = 0; r < 8; ++r) {
      const int row = rowBase + r;
      const size_t mi = (size_t)row * N + col;
      float mo = mem[mi];
      float reset = (mo - thrc) > 0.f ? thrc : 0.f;   // from previous mem
      float mn = betac * mo + (acc[nt][r] + bcol) - reset;
      mem[mi] = mn;
      float s = (mn - thrc) > 0.f ? 1.f : 0.f;
      if (F32OUT) {
        if (col < outN) spkF[(size_t)row * outN + col] = s;
      } else {
        spkH[mi] = (_Float16)s;
      }
    }
  }
}

// ---------------------------------------------------------------
extern "C" void kernel_launch(void* const* d_in, const int* in_sizes, int n_in,
                              void* d_out, int out_size, void* d_ws, size_t ws_size,
                              hipStream_t stream) {
  const float* x     = (const float*)d_in[0];
  const float* W_in  = (const float*)d_in[1];
  const float* b_in  = (const float*)d_in[2];
  const float* beta1 = (const float*)d_in[3];
  const float* thr1  = (const float*)d_in[4];
  const float* W_hid = (const float*)d_in[5];
  const float* b_hid = (const float*)d_in[6];
  const float* beta2 = (const float*)d_in[7];
  const float* thr2  = (const float*)d_in[8];
  const float* W_out = (const float*)d_in[9];
  const float* b_out = (const float*)d_in[10];
  const float* beta3 = (const float*)d_in[11];
  float* out = (float*)d_out;

  char* ws = (char*)d_ws;
  _Float16* Wi16 = (_Float16*)(ws + 0);          // 2 MB
  _Float16* Wh16 = (_Float16*)(ws + 2097152);    // 1 MB
  _Float16* Wo16 = (_Float16*)(ws + 3145728);    // 32 KB (padded 32x512)
  _Float16* s1   = (_Float16*)(ws + 3178496);    // 1 MB
  _Float16* s2   = (_Float16*)(ws + 4227072);    // 0.5 MB
  float*    mem1 = (float*)(ws + 4751360);       // 2 MB
  float*    mem2 = (float*)(ws + 6848512);       // 1 MB
  float*    mem3 = (float*)(ws + 7897088);       // 64 KB (padded N=32)
  float*    b3p    = (float*)(ws + 7962624);
  float*    beta3p = (float*)(ws + 7962752);
  float*    thr3p  = (float*)(ws + 7962880);

  snn_prep_kernel<<<4096, 256, 0, stream>>>(
      W_in, W_hid, W_out, b_out, beta3,
      Wi16, Wh16, Wo16, b3p, beta3p, thr3p, mem1, mem2, mem3);

  constexpr int T = 64, B = 512, H1 = 1024, H2 = 512, NC = 18;
  for (int t = 0; t < T; ++t) {
    // layer 1: [512x1024] = f16(x[:,t]) @ Wi^T. TDM-staged f32 A
    // (batch stride 65536) + one-time LDS f16 conversion. 1024 waves.
    snn_gemm_lif_kernel<2, false, true, true, 1024><<<256, 128, 0, stream>>>(
        nullptr, x + (size_t)t * 1024, Wi16, b_in, beta1, thr1,
        mem1, s1, nullptr, B, H1, 65536, 0);

    // layer 2: [512x512] = s1 @ Wh^T. TDM-staged f16 A. 512 waves.
    snn_gemm_lif_kernel<2, false, false, true, 1024><<<128, 128, 0, stream>>>(
        s1, nullptr, Wh16, b_hid, beta2, thr2,
        mem2, s2, nullptr, B, H2, 1024, 0);

    // layer 3: [512x32(pad)] = s2 @ Wo^T ; 32 waves, f32 out
    snn_gemm_lif_kernel<2, true, false, false, 512><<<8, 128, 0, stream>>>(
        s2, nullptr, Wo16, b3p, beta3p, thr3p,
        mem3, nullptr, out + (size_t)t * B * NC, B, 32, 512, NC);
  }
}
